// KroneckerAttention_75582834475009
// MI455X (gfx1250) — matrix operand order repaired and checked by reading, not compiled, not run on hardware
//
#include <hip/hip_runtime.h>
#include <hip/hip_bf16.h>
#include <math.h>
#include <stdint.h>

// -----------------------------------------------------------------------------
// Kronecker attention for MI455X (gfx1250, wave32, WMMA, async-to-LDS).
//
// Phase 0 (ws permitting): K_gp -> bf16 [bh][2048][64]; V_gp -> bf16 tile-major
//   transposed [bh][kt][col(256)][k(64)], so the hot loop streams bf16 tiles
//   straight into LDS with global_load_async_to_lds_b128 (ASYNCcnt tracked,
//   double-buffered: tile kt+1 streams while tile kt computes).
// Phase 1: group-mix weights wfac[bh][mi][g] = norm_g / sum_g norm_g.
// Phase 2: flash attention (Nq=Nk=2048, Dqk=64, Dv=256) per (b,h), 64 keys per
//   iteration: 8 S-WMMAs + 2 row-sum WMMAs (all-ones B carries the exact
//   flash-attention l recurrence into every lane, no shuffle reduction) +
//   32 PV-WMMAs. Group mixing folded into the epilogue.
// -----------------------------------------------------------------------------

typedef __attribute__((ext_vector_type(16))) __bf16 v16bf;
typedef __attribute__((ext_vector_type(2)))  __bf16 v2bf;
typedef __attribute__((ext_vector_type(8)))  float  v8f;

#define BH_TOTAL 32
#define NSEQ     8192
#define DHEAD    64
#define PLEN     2048
#define BM       128          // query rows per block (8 waves x 16)
#define BN       64           // keys per K-tile
#define NKT      (PLEN / BN)  // 32
#define NQT      (PLEN / BM)  // 16

// workspace layout (bytes)
#define WS_WFAC_OFF  0
#define WS_K_OFF     4096
#define WS_K_BYTES   ((size_t)BH_TOTAL * PLEN * DHEAD * 2)      // 8 MB
#define WS_V_OFF     (WS_K_OFF + WS_K_BYTES)
#define WS_V_BYTES   ((size_t)BH_TOTAL * NKT * 256 * BN * 2)    // 32 MB
#define WS_NEED      (WS_V_OFF + WS_V_BYTES)

static __device__ __forceinline__ v8f wmma_bf16(v16bf a, v16bf b, v8f c) {
  return __builtin_amdgcn_wmma_f32_16x16x32_bf16(
      false, a, false, b, (short)0, c, false, false);
}

static __device__ __forceinline__ v16bf load_frag16(const __bf16* p) {
  v16bf r;
#pragma unroll
  for (int e = 0; e < 16; ++e) r[e] = p[e];
  return r;
}

// 16B global -> LDS, tracked by ASYNCcnt (CDNA5 async-to-LDS path).
static __device__ __forceinline__ void async_b128(uint32_t lds_off,
                                                  const void* gptr) {
  asm volatile("global_load_async_to_lds_b128 %0, %1, off"
               :: "v"(lds_off), "v"((uint64_t)(uintptr_t)gptr)
               : "memory");
}

// ---------------------------------------------------------------------------
// Phase 0: bf16 conversion + V transpose into workspace. grid=(NKT, BH), 256.
// ---------------------------------------------------------------------------
__global__ __launch_bounds__(256) void convert_kv_kernel(
    const float* __restrict__ k, const float* __restrict__ v,
    __bf16* __restrict__ kws, __bf16* __restrict__ vws) {
  const int kt = blockIdx.x, bh = blockIdx.y, tid = threadIdx.x;

  {  // K tile: straight convert, row-major [key][d] (tile = 8 KB contiguous)
    const int row = tid >> 2, d0 = (tid & 3) * 16;
    const float* src =
        k + ((size_t)bh * NSEQ + 4096 + kt * BN + row) * DHEAD + d0;
    __bf16* dst = kws + ((size_t)bh * PLEN + kt * BN + row) * DHEAD + d0;
#pragma unroll
    for (int j = 0; j < 16; ++j) dst[j] = (__bf16)src[j];
  }

  // V tile: transpose to [col(256)][k(64)] via LDS, coalesced write-out.
  __shared__ __align__(64) __bf16 t[256 * BN];  // 32 KB
  {
    const int kp = tid & 31, cb = tid >> 5;     // 32 k-pairs x 8 col-blocks(32)
    const int g = cb >> 1, d0 = (cb & 1) * 32;
    const float* s0p = v + ((size_t)bh * NSEQ + (size_t)g * PLEN + kt * BN + 2 * kp) * DHEAD + d0;
    const float* s1p = s0p + DHEAD;
#pragma unroll
    for (int j = 0; j < 32; ++j) {
      v2bf pr;
      pr[0] = (__bf16)s0p[j];
      pr[1] = (__bf16)s1p[j];
      *(v2bf*)&t[(cb * 32 + j) * BN + 2 * kp] = pr;
    }
  }
  __syncthreads();
  {
    __bf16* dst = vws + (((size_t)bh * NKT + kt) * 256 + tid) * BN;
    const __bf16* s = &t[tid * BN];
#pragma unroll
    for (int j = 0; j < BN; ++j) dst[j] = s[j];  // 8x b128
  }
}

// ---------------------------------------------------------------------------
// Phase 1: group weights. One block per (b,h).
// ---------------------------------------------------------------------------
__global__ __launch_bounds__(256) void kron_weights_kernel(
    const float* __restrict__ q, const float* __restrict__ k,
    float* __restrict__ wfac) {
  const int bh = blockIdx.x, tid = threadIdx.x;
  __shared__ float ksamp[4][DHEAD];
  __shared__ float norm2[4][4];

  if (tid < 16) norm2[tid >> 2][tid & 3] = 0.f;
  {
    const int g = tid / DHEAD, d = tid % DHEAD;
    ksamp[g][d] = k[((size_t)bh * NSEQ + (size_t)g * PLEN) * DHEAD + d];
  }
  __syncthreads();

  for (int row = tid; row < NSEQ; row += 256) {
    const float* qr = q + ((size_t)bh * NSEQ + row) * DHEAD;
    float acc[4] = {0.f, 0.f, 0.f, 0.f};
    for (int d = 0; d < DHEAD; ++d) {
      const float qv = qr[d];
#pragma unroll
      for (int g = 0; g < 4; ++g) acc[g] = fmaf(qv, ksamp[g][d], acc[g]);
    }
    const int mi = row >> 11;
#pragma unroll
    for (int g = 0; g < 4; ++g) atomicAdd(&norm2[mi][g], acc[g] * acc[g]);
  }
  __syncthreads();

  if (tid < 4) {
    float nv[4], s = 0.f;
#pragma unroll
    for (int g = 0; g < 4; ++g) { nv[g] = sqrtf(norm2[tid][g]); s += nv[g]; }
    const float inv = 1.f / s;
#pragma unroll
    for (int g = 0; g < 4; ++g) wfac[bh * 16 + tid * 4 + g] = nv[g] * inv;
  }
}

// ---------------------------------------------------------------------------
// Phase 2: flash attention + group mixing. grid=(NQT, BH), 256 threads.
// ---------------------------------------------------------------------------
template <bool ASYNC>
__global__ __launch_bounds__(256) void kron_attn_kernel(
    const float* __restrict__ q, const float* __restrict__ k,
    const float* __restrict__ v, const __bf16* __restrict__ kws,
    const __bf16* __restrict__ vws, const float* __restrict__ wfac,
    float* __restrict__ out) {
  const int qt = blockIdx.x, bh = blockIdx.y, tid = threadIdx.x;
  const int wave = tid >> 5, lane = tid & 31;
  const int half = lane >> 4, lm = lane & 15;

  __shared__ __align__(64) __bf16 k_lds[2][BN * DHEAD];  // 16 KB [key][d]
  __shared__ __align__(64) __bf16 v_lds[2][256 * BN];    // 64 KB [col][k]
  __shared__ __align__(64) __bf16 p_lds[8 * 16 * BN];    // 16 KB wave-private P

  float wf[16];
#pragma unroll
  for (int i = 0; i < 16; ++i) wf[i] = wfac[bh * 16 + i];

  const float qscale = 0.125f * 1.4426950408889634f;  // 1/sqrt(D) * log2(e)

  // Q A-fragments (16x64 -> two 16x32 bf16), loaded once.
  const float* qbase =
      q + ((size_t)bh * NSEQ + 4096 + (size_t)qt * BM + wave * 16 + lm) * DHEAD;
  v16bf a_q[2];
#pragma unroll
  for (int c = 0; c < 2; ++c)
#pragma unroll
    for (int e = 0; e < 16; ++e) {
      const int d = c * 32 + half * 16 + e;
      a_q[c][e] = (__bf16)(qbase[d] * qscale);
    }

  v16bf ones;  // all-ones B: row-sum WMMAs carry the l recurrence to all lanes
#pragma unroll
  for (int e = 0; e < 16; ++e) ones[e] = (__bf16)1.0f;

  v8f o[16], o_sum;
#pragma unroll
  for (int t = 0; t < 16; ++t)
#pragma unroll
    for (int e = 0; e < 8; ++e) o[t][e] = 0.f;
#pragma unroll
  for (int e = 0; e < 8; ++e) o_sum[e] = 0.f;
  float m_run[8];
#pragma unroll
  for (int r = 0; r < 8; ++r) m_run[r] = -__builtin_inff();

  const float* kbase = k + ((size_t)bh * NSEQ + 4096) * DHEAD;
  const float* vbase = v + (size_t)bh * NSEQ * DHEAD;
  const __bf16* kwsb = kws + (size_t)bh * PLEN * DHEAD;      //  8 KB / tile
  const __bf16* vwsb = vws + (size_t)bh * NKT * 256 * BN;    // 32 KB / tile
  __bf16* pm = &p_lds[wave * 16 * BN];

  const uint32_t kl0 = (uint32_t)(uintptr_t)(void*)&k_lds[0][0];
  const uint32_t kl1 = (uint32_t)(uintptr_t)(void*)&k_lds[1][0];
  const uint32_t vl0 = (uint32_t)(uintptr_t)(void*)&v_lds[0][0];
  const uint32_t vl1 = (uint32_t)(uintptr_t)(void*)&v_lds[1][0];

  // Issue one tile's async loads (2x K + 8x V b128 per thread) into buffer b.
  auto issue_tile = [&](int kt, int b) {
    const char* kg = (const char*)(kwsb + (size_t)kt * BN * DHEAD);
    const char* vg = (const char*)(vwsb + (size_t)kt * 256 * BN);
    const uint32_t kl = (b ? kl1 : kl0) + tid * 32;
    const uint32_t vl = (b ? vl1 : vl0) + tid * 128;
#pragma unroll
    for (int j = 0; j < 2; ++j) async_b128(kl + j * 16, kg + tid * 32 + j * 16);
#pragma unroll
    for (int j = 0; j < 8; ++j) async_b128(vl + j * 16, vg + tid * 128 + j * 16);
  };

  if constexpr (ASYNC) issue_tile(0, 0);

  for (int kt = 0; kt < NKT; ++kt) {
    const int cur = kt & 1;
    const __bf16* kl = &k_lds[cur][0];
    const __bf16* vl = &v_lds[cur][0];

    if constexpr (ASYNC) {
      if (kt + 1 < NKT) {
        issue_tile(kt + 1, cur ^ 1);
        asm volatile("s_wait_asynccnt 0xa" ::: "memory");  // tile kt landed
      } else {
        asm volatile("s_wait_asynccnt 0x0" ::: "memory");
      }
    } else {
      {  // cooperative K tile: 64x64 f32 -> bf16, row-major [key][d]
        const int row = tid >> 2, d0 = (tid & 3) * 16;
        const float* src = kbase + ((size_t)(kt * BN + row)) * DHEAD + d0;
        __bf16* dst = (__bf16*)kl;
#pragma unroll
        for (int j = 0; j < 16; ++j) dst[row * DHEAD + d0 + j] = (__bf16)src[j];
      }
      {  // cooperative V tile, transposed into [col][k], pair-packed
        const int kp = tid & 31, cb = tid >> 5;
        const int g = cb >> 1, d0 = (cb & 1) * 32;
        const float* s0p =
            vbase + ((size_t)g * PLEN + kt * BN + 2 * kp) * DHEAD + d0;
        const float* s1p = s0p + DHEAD;
#pragma unroll
        for (int j = 0; j < 32; ++j) {
          v2bf pr;
          pr[0] = (__bf16)s0p[j];
          pr[1] = (__bf16)s1p[j];
          *(v2bf*)&((__bf16*)vl)[(cb * 32 + j) * BN + 2 * kp] = pr;
        }
      }
    }
    __syncthreads();

    // ---- S = Q*K^T (scaled, log2 domain): four 16x16 fragments (64 keys) ----
    v8f s[4];
#pragma unroll
    for (int nh = 0; nh < 4; ++nh) {
      v8f acc;
#pragma unroll
      for (int e = 0; e < 8; ++e) acc[e] = 0.f;
#pragma unroll
      for (int c = 0; c < 2; ++c) {
        v16bf bk =
            load_frag16(&kl[(nh * 16 + lm) * DHEAD + c * 32 + half * 16]);
        acc = wmma_bf16(a_q[c], bk, acc);
      }
      s[nh] = acc;
    }

    // ---- online softmax max (row = r + 8*half, column = lm) ----
    float mx[8];
#pragma unroll
    for (int r = 0; r < 8; ++r)
      mx[r] = fmaxf(fmaxf(s[0][r], s[1][r]), fmaxf(s[2][r], s[3][r]));
#pragma unroll
    for (int mask = 1; mask < 16; mask <<= 1)
#pragma unroll
      for (int r = 0; r < 8; ++r)
        mx[r] = fmaxf(mx[r], __shfl_xor(mx[r], mask, 32));

    float al[8];
#pragma unroll
    for (int r = 0; r < 8; ++r) {
      const float mn = fmaxf(m_run[r], mx[r]);
      al[r] = exp2f(m_run[r] - mn);
      m_run[r] = mn;
    }
#pragma unroll
    for (int nh = 0; nh < 4; ++nh)
#pragma unroll
      for (int r = 0; r < 8; ++r) s[nh][r] = exp2f(s[nh][r] - m_run[r]);

    // rescale accumulators (o_sum carries the l recurrence)
#pragma unroll
    for (int t = 0; t < 16; ++t)
#pragma unroll
      for (int r = 0; r < 8; ++r) o[t][r] *= al[r];
#pragma unroll
    for (int r = 0; r < 8; ++r) o_sum[r] *= al[r];

    // ---- P: C-layout -> A-layout via wave-private LDS bounce ----
#pragma unroll
    for (int nh = 0; nh < 4; ++nh)
#pragma unroll
      for (int r = 0; r < 8; ++r)
        pm[(r + 8 * half) * BN + nh * 16 + lm] = (__bf16)s[nh][r];
    v16bf a_p[2];
#pragma unroll
    for (int h = 0; h < 2; ++h)
#pragma unroll
      for (int e = 0; e < 16; ++e) {
        const int kk = half * 16 + e;
        a_p[h][e] = pm[lm * BN + h * 32 + kk];
      }

    // ---- row sums (l) via ones-B WMMAs ----
    o_sum = wmma_bf16(a_p[0], ones, o_sum);
    o_sum = wmma_bf16(a_p[1], ones, o_sum);

    // ---- O += P * V over 256 value columns (32 WMMAs) ----
#pragma unroll
    for (int h = 0; h < 2; ++h)
#pragma unroll
      for (int t = 0; t < 16; ++t) {
        v16bf bv = load_frag16(&vl[(t * 16 + lm) * BN + h * 32 + half * 16]);
        o[t] = wmma_bf16(a_p[h], bv, o[t]);
      }
    __syncthreads();  // protect LDS buffers before reuse
  }

  // ---- epilogue: out[mi*2048+j, d] = sum_g O[j,g*64+d] * wf[mi][g] / l ----
  const int jbase = qt * BM + wave * 16;
#pragma unroll
  for (int r = 0; r < 8; ++r) {
    const int row = jbase + r + 8 * half;
    const float invl = 1.f / o_sum[r];
#pragma unroll
    for (int mi = 0; mi < 4; ++mi) {
#pragma unroll
      for (int ds = 0; ds < 4; ++ds) {
        float acc = 0.f;
#pragma unroll
        for (int g = 0; g < 4; ++g)
          acc = fmaf(o[g * 4 + ds][r], wf[mi * 4 + g], acc);
        float* dst = out +
                     ((size_t)bh * NSEQ + (size_t)mi * PLEN + row) * DHEAD +
                     ds * 16 + lm;
        __builtin_nontemporal_store(acc * invl, dst);
      }
    }
  }
}

// ---------------------------------------------------------------------------
extern "C" void kernel_launch(void* const* d_in, const int* in_sizes, int n_in,
                              void* d_out, int out_size, void* d_ws, size_t ws_size,
                              hipStream_t stream) {
  const float* q = (const float*)d_in[0];
  const float* k = (const float*)d_in[1];
  const float* v = (const float*)d_in[2];
  float* out  = (float*)d_out;
  float* wfac = (float*)((char*)d_ws + WS_WFAC_OFF);

  kron_weights_kernel<<<BH_TOTAL, 256, 0, stream>>>(q, k, wfac);

  if (ws_size >= WS_NEED) {
    __bf16* kws = (__bf16*)((char*)d_ws + WS_K_OFF);
    __bf16* vws = (__bf16*)((char*)d_ws + WS_V_OFF);
    convert_kv_kernel<<<dim3(NKT, BH_TOTAL), 256, 0, stream>>>(k, v, kws, vws);
    kron_attn_kernel<true><<<dim3(NQT, BH_TOTAL), 256, 0, stream>>>(
        q, k, v, kws, vws, wfac, out);
  } else {
    kron_attn_kernel<false><<<dim3(NQT, BH_TOTAL), 256, 0, stream>>>(
        q, k, v, nullptr, nullptr, wfac, out);
  }
}
